// GNN_87960930222404
// MI455X (gfx1250) — compile-verified, matrix operand order checked by reading, not run on hardware
//
#include <hip/hip_runtime.h>
#include <hip/hip_bf16.h>

typedef __attribute__((ext_vector_type(16))) _Float16 v16h;
typedef __attribute__((ext_vector_type(8)))  float    v8f;

union AF { v16h v; uint4 q[2]; };

struct MLP4P {
    const _Float16* wf[4];
    const float* b[4];
    const float* g[4];
    const float* be[4];
};

// ---- fragment loaders -------------------------------------------------------
// A-matrix 16x32 f16 layout (ISA 7.12.2): lane m=lane&15, half=lane>>4 holds
// K = [k0+8*half .. +7] and [k0+16+8*half .. +7]  -> two contiguous 16B runs.
__device__ __forceinline__ v16h load_frag_a(const _Float16* base, int ld, int k0, int lane) {
    const int m = lane & 15, half = lane >> 4;
    const _Float16* p = base + m * ld + k0 + half * 8;
    AF a;
    a.q[0] = *(const uint4*)(p);
    a.q[1] = *(const uint4*)(p + 16);
    return a.v;
}

// B fragments are pre-swizzled: frag-major, 32 lanes x 16 f16 contiguous.
__device__ __forceinline__ v16h load_frag_b(const _Float16* wf, int frag, int lane) {
    const _Float16* p = wf + ((size_t)frag * 32 + lane) * 16;
    AF a;
    a.q[0] = ((const uint4*)p)[0];
    a.q[1] = ((const uint4*)p)[1];
    return a.v;
}

// ---- one Linear(+opt LayerNorm)+ReLU layer, N=64 out, per-wave 16-row tile --
template<int KT, bool LN>
__device__ __forceinline__ void mlp_layer(const _Float16* aIn, _Float16* aOut,
                                          const _Float16* wf, const float* bias,
                                          const float* gamma, const float* beta,
                                          int lane, v8f c[4])
{
    const int n16 = lane & 15, half = lane >> 4;
    v16h a[KT];
#pragma unroll
    for (int kt = 0; kt < KT; ++kt) a[kt] = load_frag_a(aIn, KT * 32, kt * 32, lane);

#pragma unroll
    for (int nt = 0; nt < 4; ++nt) {
        v8f acc;
#pragma unroll
        for (int r = 0; r < 8; ++r) acc[r] = 0.0f;
#pragma unroll
        for (int kt = 0; kt < KT; ++kt) {
            v16h b = load_frag_b(wf, kt * 4 + nt, lane);
            acc = __builtin_amdgcn_wmma_f32_16x16x32_f16(false, a[kt], false, b,
                                                         (short)0, acc, false, false);
        }
        c[nt] = acc;
    }

    float bn[4];
#pragma unroll
    for (int nt = 0; nt < 4; ++nt) bn[nt] = bias[nt * 16 + n16];
#pragma unroll
    for (int nt = 0; nt < 4; ++nt)
#pragma unroll
        for (int r = 0; r < 8; ++r) c[nt][r] += bn[nt];

    if (LN) {
        float gn[4], ben[4];
#pragma unroll
        for (int nt = 0; nt < 4; ++nt) { gn[nt] = gamma[nt * 16 + n16]; ben[nt] = beta[nt * 16 + n16]; }
#pragma unroll
        for (int r = 0; r < 8; ++r) {
            // row (r + 8*half): its 64 values sit in 16 lanes of this half x 4 nt frags
            float s = 0.0f, s2 = 0.0f;
#pragma unroll
            for (int nt = 0; nt < 4; ++nt) { float v = c[nt][r]; s += v; s2 += v * v; }
#pragma unroll
            for (int m = 1; m <= 8; m <<= 1) { s += __shfl_xor(s, m); s2 += __shfl_xor(s2, m); }
            float mean = s * (1.0f / 64.0f);
            float var  = s2 * (1.0f / 64.0f) - mean * mean;
            float rstd = rsqrtf(var + 1e-5f);
#pragma unroll
            for (int nt = 0; nt < 4; ++nt) {
                float v = (c[nt][r] - mean) * rstd * gn[nt] + ben[nt];
                c[nt][r] = fmaxf(v, 0.0f);
            }
        }
    } else {
#pragma unroll
        for (int nt = 0; nt < 4; ++nt)
#pragma unroll
            for (int r = 0; r < 8; ++r) c[nt][r] = fmaxf(c[nt][r], 0.0f);
    }

    if (aOut) {
        // C/D layout: VGPR r = row r+8*half, col = lane&15  -> write A-layout LDS
#pragma unroll
        for (int nt = 0; nt < 4; ++nt)
#pragma unroll
            for (int r = 0; r < 8; ++r)
                aOut[(r + 8 * half) * 64 + nt * 16 + n16] = (_Float16)c[nt][r];
    }
}

// ---- weight fp32 -> fp16 pre-swizzle into B-fragment order ------------------
__global__ void gnn_swizzle_w(const float* __restrict__ W, _Float16* __restrict__ dst,
                              int Ksrc, int Ntiles, int total)
{
    int t = blockIdx.x * 256 + threadIdx.x;
    if (t >= total) return;
    int j = t & 15;
    int lane = (t >> 4) & 31;
    int f = t >> 9;
    int kt = f / Ntiles, nt = f % Ntiles;
    int half = lane >> 4, n = lane & 15;
    int i = j >> 1, e = j & 1;
    int klocal = ((i & 3) * 2 + e) + ((i >> 2) * 16) + half * 8;
    int k = kt * 32 + klocal;
    int N = Ntiles * 16;
    float v = (k < Ksrc) ? W[(size_t)k * N + nt * 16 + n] : 0.0f;
    dst[t] = (_Float16)v;
}

// ---- degree = counts of node id in concat(ei1, ei0) -------------------------
__global__ void gnn_deg_zero(float* deg, int Nn) {
    int t = blockIdx.x * 256 + threadIdx.x;
    if (t < Nn) deg[t] = 0.0f;
}
__global__ void gnn_deg_count(const int* __restrict__ ei, float* deg, int E) {
    int t = blockIdx.x * 256 + threadIdx.x;
    if (t < E) {
        atomicAdd(&deg[ei[t]], 1.0f);
        atomicAdd(&deg[ei[E + t]], 1.0f);
    }
}

// ---- encoder: x[N,6] -> 64 -> 64, ReLU, no LN -------------------------------
__global__ __launch_bounds__(128) void gnn_enc_kernel(const float* __restrict__ x,
        const _Float16* __restrict__ wf0, const _Float16* __restrict__ wf1,
        const float* __restrict__ b0, const float* __restrict__ b1,
        _Float16* __restrict__ hf, int Nn)
{
    __shared__ __align__(16) _Float16 sIn[4][16 * 32];
    __shared__ __align__(16) _Float16 sMid[4][16 * 64];
    int wave = threadIdx.x >> 5, lane = threadIdx.x & 31;
    int rowBase = (blockIdx.x * 4 + wave) * 16;
#pragma unroll
    for (int i = 0; i < 16; ++i) sIn[wave][i * 32 + lane] = (_Float16)0.0f;
    if (lane < 16) {
        int row = rowBase + lane;
        int rr = row < Nn ? row : 0;
#pragma unroll
        for (int cidx = 0; cidx < 6; ++cidx)
            sIn[wave][lane * 32 + cidx] = (_Float16)x[(size_t)rr * 6 + cidx];
    }
    v8f c[4];
    mlp_layer<1, false>(sIn[wave], sMid[wave], wf0, b0, nullptr, nullptr, lane, c);
    mlp_layer<2, false>(sMid[wave], nullptr, wf1, b1, nullptr, nullptr, lane, c);
    int n16 = lane & 15, half = lane >> 4;
#pragma unroll
    for (int r = 0; r < 8; ++r) {
        int row = rowBase + r + 8 * half;
        if (row < Nn) {
#pragma unroll
            for (int nt = 0; nt < 4; ++nt)
                hf[(size_t)row * 64 + nt * 16 + n16] = (_Float16)c[nt][r];
        }
    }
}

// ---- edge net on rows 0..N-1 only; agg[n] = deg[n] * e[n] -------------------
__global__ __launch_bounds__(128) void gnn_edge_kernel(const _Float16* __restrict__ hf,
        const int* __restrict__ ei, const float* __restrict__ deg,
        MLP4P P, _Float16* __restrict__ agg, int Nn, int Ecnt)
{
    __shared__ __align__(16) _Float16 sIn[4][16 * 128];
    __shared__ __align__(16) _Float16 sMid[4][16 * 64];
    int wave = threadIdx.x >> 5, lane = threadIdx.x & 31;
    int rowBase = (blockIdx.x * 4 + wave) * 16;
    int n16 = lane & 15, half = lane >> 4;
    for (int r = 0; r < 16; ++r) {
        int row = rowBase + r;
        int rr = row < Nn ? row : 0;
        int node = half ? ei[Ecnt + rr] : ei[rr];   // cols 0..63 <- h[send], 64..127 <- h[recv]
        uint2 v = *(const uint2*)(hf + (size_t)node * 64 + n16 * 4);
        *(uint2*)&sIn[wave][r * 128 + half * 64 + n16 * 4] = v;
    }
    v8f c[4];
    mlp_layer<4, true>(sIn[wave], sMid[wave], P.wf[0], P.b[0], P.g[0], P.be[0], lane, c);
    mlp_layer<2, true>(sMid[wave], sIn[wave], P.wf[1], P.b[1], P.g[1], P.be[1], lane, c);
    mlp_layer<2, true>(sIn[wave], sMid[wave], P.wf[2], P.b[2], P.g[2], P.be[2], lane, c);
    mlp_layer<2, true>(sMid[wave], nullptr,   P.wf[3], P.b[3], P.g[3], P.be[3], lane, c);
#pragma unroll
    for (int r = 0; r < 8; ++r) {
        int row = rowBase + r + 8 * half;
        if (row < Nn) {
            float d = deg[row];
#pragma unroll
            for (int nt = 0; nt < 4; ++nt)
                agg[(size_t)row * 64 + nt * 16 + n16] = (_Float16)(c[nt][r] * d);
        }
    }
}

// ---- node net: [h, agg] -> 4 LN layers -> + h0 ------------------------------
__global__ __launch_bounds__(128) void gnn_node_kernel(const _Float16* __restrict__ hf,
        const _Float16* __restrict__ agg, MLP4P P, _Float16* __restrict__ hnew, int Nn)
{
    __shared__ __align__(16) _Float16 sIn[4][16 * 128];
    __shared__ __align__(16) _Float16 sMid[4][16 * 64];
    int wave = threadIdx.x >> 5, lane = threadIdx.x & 31;
    int rowBase = (blockIdx.x * 4 + wave) * 16;
    int n16 = lane & 15, half = lane >> 4;
    for (int r = 0; r < 16; ++r) {
        int row = rowBase + r;
        int rr = row < Nn ? row : 0;
        const _Float16* src = (half ? agg : hf) + (size_t)rr * 64 + n16 * 4;
        *(uint2*)&sIn[wave][r * 128 + half * 64 + n16 * 4] = *(const uint2*)src;
    }
    v8f c[4];
    mlp_layer<4, true>(sIn[wave], sMid[wave], P.wf[0], P.b[0], P.g[0], P.be[0], lane, c);
    mlp_layer<2, true>(sMid[wave], sIn[wave], P.wf[1], P.b[1], P.g[1], P.be[1], lane, c);
    mlp_layer<2, true>(sIn[wave], sMid[wave], P.wf[2], P.b[2], P.g[2], P.be[2], lane, c);
    mlp_layer<2, true>(sMid[wave], nullptr,   P.wf[3], P.b[3], P.g[3], P.be[3], lane, c);
#pragma unroll
    for (int r = 0; r < 8; ++r) {
        int row = rowBase + r + 8 * half;
        if (row < Nn) {
#pragma unroll
            for (int nt = 0; nt < 4; ++nt) {
                float h0 = (float)hf[(size_t)row * 64 + nt * 16 + n16];
                hnew[(size_t)row * 64 + nt * 16 + n16] = (_Float16)(c[nt][r] + h0);
            }
        }
    }
}

// ---- classifier: [h[s], h[t]] -> 64 (ReLU) -> 1 -----------------------------
__global__ __launch_bounds__(128) void gnn_clf_kernel(const _Float16* __restrict__ hn,
        const int* __restrict__ ei, const _Float16* __restrict__ wf0,
        const float* __restrict__ b0, const float* __restrict__ w2,
        const float* __restrict__ b2, float* __restrict__ out, int Ecnt)
{
    __shared__ __align__(16) _Float16 sIn[4][16 * 128];
    int wave = threadIdx.x >> 5, lane = threadIdx.x & 31;
    int rowBase = (blockIdx.x * 4 + wave) * 16;
    int n16 = lane & 15, half = lane >> 4;
    for (int r = 0; r < 16; ++r) {
        int row = rowBase + r;
        int rr = row < Ecnt ? row : 0;
        int node = half ? ei[Ecnt + rr] : ei[rr];   // s then t
        uint2 v = *(const uint2*)(hn + (size_t)node * 64 + n16 * 4);
        *(uint2*)&sIn[wave][r * 128 + half * 64 + n16 * 4] = v;
    }
    v8f c[4];
    mlp_layer<4, false>(sIn[wave], nullptr, wf0, b0, nullptr, nullptr, lane, c);
    float w2n[4];
#pragma unroll
    for (int nt = 0; nt < 4; ++nt) w2n[nt] = w2[nt * 16 + n16];
    float p[8];
#pragma unroll
    for (int r = 0; r < 8; ++r) {
        float s = 0.0f;
#pragma unroll
        for (int nt = 0; nt < 4; ++nt) s += c[nt][r] * w2n[nt];
#pragma unroll
        for (int m = 1; m <= 8; m <<= 1) s += __shfl_xor(s, m);
        p[r] = s;
    }
    if (n16 == 0) {
        float bb = b2[0];
#pragma unroll
        for (int r = 0; r < 8; ++r) {
            int row = rowBase + r + 8 * half;
            if (row < Ecnt) out[row] = p[r] + bb;
        }
    }
}

// ---- host orchestration -----------------------------------------------------
extern "C" void kernel_launch(void* const* d_in, const int* in_sizes, int n_in,
                              void* d_out, int out_size, void* d_ws, size_t ws_size,
                              hipStream_t stream)
{
    const int N_ = in_sizes[0] / 6;     // nodes
    const int E_ = in_sizes[1] / 2;     // directed edges

    const float* x     = (const float*)d_in[0];
    const int*   ei    = (const int*)d_in[1];
    const float* encW0 = (const float*)d_in[2];
    const float* encW1 = (const float*)d_in[3];
    const float* encB0 = (const float*)d_in[4];
    const float* encB1 = (const float*)d_in[5];
    const float* edgeW[4], *edgeB[4], *edgeG[4], *edgeBe[4];
    const float* nodeW[4], *nodeB[4], *nodeG[4], *nodeBe[4];
    for (int i = 0; i < 4; ++i) {
        edgeW[i]  = (const float*)d_in[6 + i];
        edgeB[i]  = (const float*)d_in[10 + i];
        edgeG[i]  = (const float*)d_in[14 + i];
        edgeBe[i] = (const float*)d_in[18 + i];
        nodeW[i]  = (const float*)d_in[22 + i];
        nodeB[i]  = (const float*)d_in[26 + i];
        nodeG[i]  = (const float*)d_in[30 + i];
        nodeBe[i] = (const float*)d_in[34 + i];
    }
    const float* clfW0 = (const float*)d_in[38];
    const float* clfW1 = (const float*)d_in[39];
    const float* clfB0 = (const float*)d_in[40];
    const float* clfB1 = (const float*)d_in[41];

    char* ws = (char*)d_ws;
    size_t off = 0;
    auto alloc = [&](size_t bytes) -> char* {
        char* p = ws + off;
        off = (off + bytes + 255) & ~(size_t)255;
        return p;
    };
    _Float16* wEnc0 = (_Float16*)alloc(2048 * 2);   // Kpad=32, N=64 : 1x4 frags
    _Float16* wEnc1 = (_Float16*)alloc(4096 * 2);   // K=64        : 2x4 frags
    _Float16* wE[4], *wN[4];
    wE[0] = (_Float16*)alloc(8192 * 2);             // K=128       : 4x4 frags
    for (int i = 1; i < 4; ++i) wE[i] = (_Float16*)alloc(4096 * 2);
    wN[0] = (_Float16*)alloc(8192 * 2);
    for (int i = 1; i < 4; ++i) wN[i] = (_Float16*)alloc(4096 * 2);
    _Float16* wC0  = (_Float16*)alloc(8192 * 2);
    _Float16* hf   = (_Float16*)alloc((size_t)N_ * 64 * 2);
    _Float16* agg  = (_Float16*)alloc((size_t)N_ * 64 * 2);
    _Float16* hnew = (_Float16*)alloc((size_t)N_ * 64 * 2);
    float*    deg  = (float*)alloc((size_t)N_ * 4);

    auto sw = [&](const float* W, _Float16* dst, int Ksrc, int Kpad, int N) {
        int total = (Kpad / 32) * (N / 16) * 512;
        gnn_swizzle_w<<<dim3((total + 255) / 256), dim3(256), 0, stream>>>(W, dst, Ksrc, N / 16, total);
    };
    sw(encW0, wEnc0, 6, 32, 64);
    sw(encW1, wEnc1, 64, 64, 64);
    sw(edgeW[0], wE[0], 128, 128, 64);
    for (int i = 1; i < 4; ++i) sw(edgeW[i], wE[i], 64, 64, 64);
    sw(nodeW[0], wN[0], 128, 128, 64);
    for (int i = 1; i < 4; ++i) sw(nodeW[i], wN[i], 64, 64, 64);
    sw(clfW0, wC0, 128, 128, 64);

    gnn_deg_zero<<<dim3((N_ + 255) / 256), dim3(256), 0, stream>>>(deg, N_);
    gnn_deg_count<<<dim3((E_ + 255) / 256), dim3(256), 0, stream>>>(ei, deg, E_);

    int blocksN = (N_ + 63) / 64;
    gnn_enc_kernel<<<dim3(blocksN), dim3(128), 0, stream>>>(x, wEnc0, wEnc1, encB0, encB1, hf, N_);

    MLP4P PE, PN;
    for (int i = 0; i < 4; ++i) {
        PE.wf[i] = wE[i]; PE.b[i] = edgeB[i]; PE.g[i] = edgeG[i]; PE.be[i] = edgeBe[i];
        PN.wf[i] = wN[i]; PN.b[i] = nodeB[i]; PN.g[i] = nodeG[i]; PN.be[i] = nodeBe[i];
    }
    gnn_edge_kernel<<<dim3(blocksN), dim3(128), 0, stream>>>(hf, ei, deg, PE, agg, N_, E_);
    gnn_node_kernel<<<dim3(blocksN), dim3(128), 0, stream>>>(hf, agg, PN, hnew, N_);

    int blocksE = (E_ + 63) / 64;
    gnn_clf_kernel<<<dim3(blocksE), dim3(128), 0, stream>>>(hnew, ei, wC0, clfB0, clfW1, clfB1,
                                                            (float*)d_out, E_);
}
